// CustomGRUCell_70523363000807
// MI455X (gfx1250) — matrix operand order repaired and checked by reading, not compiled, not run on hardware
//
#include <hip/hip_runtime.h>
#include <hip/hip_bf16.h>
#include <math.h>

#if defined(__has_builtin)
#if __has_builtin(__builtin_amdgcn_global_load_async_to_lds_b128) && \
    __has_builtin(__builtin_amdgcn_s_wait_asynccnt)
#define USE_ASYNC 1
#else
#define USE_ASYNC 0
#endif
#else
#define USE_ASYNC 0
#endif

typedef __attribute__((ext_vector_type(16))) __bf16 v16bf;
typedef __attribute__((ext_vector_type(8)))  __bf16 v8bf;
typedef __attribute__((ext_vector_type(8)))  float  v8f;
typedef __attribute__((ext_vector_type(4)))  int    v4i;

#if USE_ASYNC
typedef __attribute__((address_space(1))) v4i* gv4ip;   // global int4*
typedef __attribute__((address_space(3))) v4i* lv4ip;   // LDS int4*
#define ASYNC_CP(g, l) \
  __builtin_amdgcn_global_load_async_to_lds_b128((gv4ip)(g), (lv4ip)(l), 0, 0)
#endif

#define BDIM 16384
#define DDIM 512
#define UDIM 512
#define KDIM 1024      /* D + U */
#define NDIM 512
#define BK   32
#define LDP  40        /* LDS row stride in halves (padded, conflict-friendly) */
#define NKT  (KDIM / BK)

__device__ __forceinline__ v16bf cat8(v8bf lo, v8bf hi) {
  return __builtin_shufflevector(lo, hi, 0,1,2,3,4,5,6,7,8,9,10,11,12,13,14,15);
}
__device__ __forceinline__ float sigmoidf_(float x) { return 1.0f / (1.0f + __expf(-x)); }

// ---------------- prep: x_h = concat(x, h) as bf16 [B][K] ----------------
__global__ __launch_bounds__(256)
void pack_xh(const float* __restrict__ x, const float* __restrict__ h,
             __bf16* __restrict__ A) {
  size_t h8 = ((size_t)blockIdx.x * 256 + threadIdx.x) * 8;   // 8 halves per thread
  int b = (int)(h8 >> 10);
  int c = (int)(h8 & 1023);
  const float* src = (c < DDIM) ? (x + (size_t)b * DDIM + c)
                                : (h + (size_t)b * UDIM + (c - DDIM));
  v8bf o;
#pragma unroll
  for (int i = 0; i < 8; ++i) o[i] = (__bf16)src[i];
  *(v8bf*)(A + h8) = o;
}

// ---------------- prep: WT[n][k] = bf16(W[k][n]) for 3 weight mats ----------------
__global__ __launch_bounds__(256)
void conv_wt(const float* __restrict__ Wz, const float* __restrict__ Wr,
             const float* __restrict__ Wh,
             __bf16* __restrict__ WzT, __bf16* __restrict__ WrT, __bf16* __restrict__ WhT) {
  const float* W = (blockIdx.y == 0) ? Wz : (blockIdx.y == 1) ? Wr : Wh;
  __bf16*     WT = (blockIdx.y == 0) ? WzT : (blockIdx.y == 1) ? WrT : WhT;
  size_t idx = (size_t)blockIdx.x * 256 + threadIdx.x;  // over KDIM*NDIM, coalesced read
  int k = (int)(idx / NDIM);
  int n = (int)(idx % NDIM);
  WT[(size_t)n * KDIM + k] = (__bf16)W[idx];
}

// ---------------- fused GEMM + gate epilogue ----------------
// GATE==0: grid.z picks Wz (z = sigmoid -> zout) or Wr (rh = bf16(sigmoid*h) -> RH_out)
// GATE==2: h_hat GEMM over [x | r*h], then out = z*h + (1-z)*tanh(.)  (zout doubles as out)
template<int GATE>
__global__ __launch_bounds__(256)
void gru_gemm(const __bf16* __restrict__ Abuf, const __bf16* __restrict__ RH_in,
              const __bf16* __restrict__ WT0, const __bf16* __restrict__ WT1,
              const float* __restrict__ h_prev,
              float* zout, __bf16* RH_out) {
  __shared__ __bf16 ldsA[2][128 * LDP];
  __shared__ __bf16 ldsB[2][128 * LDP];

  const int t    = threadIdx.x;
  const int lane = t & 31;
  const int wid  = t >> 5;
  const int wr   = wid >> 2;            // 0..1  -> 64-row strip
  const int wc   = wid & 3;             // 0..3  -> 32-col strip
  const int l16  = lane & 15;
  const int hs   = (lane >> 4) * 8;     // A chunk select: K{0-7,16-23} vs K{8-15,24-31}
  const int bks  = (lane >> 4) * 16;    // unused; B selects K chunks with hs, same as A

  const int mBase = blockIdx.y * 128;
  const int nBase = blockIdx.x * 128;

  const __bf16* WT = WT0;
  if (GATE == 0 && blockIdx.z == 1) WT = WT1;

  const int lr = t >> 2;                // 0..63 loader row
  const int lc = (t & 3) * 8;           // loader half-offset within K-tile

  auto srcA = [&](int kt, const __bf16*& base, size_t& stride) {
    int k0 = kt * BK;
    if (GATE == 2 && k0 >= DDIM) { base = RH_in + (size_t)mBase * UDIM + (k0 - DDIM); stride = UDIM; }
    else                         { base = Abuf  + (size_t)mBase * KDIM +  k0;         stride = KDIM; }
  };

  v8f acc[4][2];
#pragma unroll
  for (int mt = 0; mt < 4; ++mt)
#pragma unroll
    for (int nt = 0; nt < 2; ++nt) acc[mt][nt] = (v8f)(0.0f);

  auto computeTile = [&](int buf) {
    v16bf af[4];
#pragma unroll
    for (int mt = 0; mt < 4; ++mt) {
      const __bf16* p = &ldsA[buf][(wr * 64 + mt * 16 + l16) * LDP];
      af[mt] = cat8(*(const v8bf*)(p + hs), *(const v8bf*)(p + 16 + hs));
    }
#pragma unroll
    for (int nt = 0; nt < 2; ++nt) {
      const __bf16* q = &ldsB[buf][(wc * 32 + nt * 16 + l16) * LDP + hs];
      v16bf bf = cat8(*(const v8bf*)(q), *(const v8bf*)(q + 16));
#pragma unroll
      for (int mt = 0; mt < 4; ++mt)
        acc[mt][nt] = __builtin_amdgcn_wmma_f32_16x16x32_bf16(
            false, af[mt], false, bf, (short)0, acc[mt][nt], false, false);
    }
  };

#if USE_ASYNC
  // ---- CDNA5 async global->LDS pipeline (ASYNCcnt), double-buffered LDS ----
  auto issueTile = [&](int kt, int buf) {
    const __bf16* abase; size_t astride;
    srcA(kt, abase, astride);
    const __bf16* bbase = WT + (size_t)nBase * KDIM + kt * BK;
    ASYNC_CP(abase + (size_t)lr        * astride + lc, &ldsA[buf][lr        * LDP + lc]);
    ASYNC_CP(abase + (size_t)(lr + 64) * astride + lc, &ldsA[buf][(lr + 64) * LDP + lc]);
    ASYNC_CP(bbase + (size_t)lr        * KDIM    + lc, &ldsB[buf][lr        * LDP + lc]);
    ASYNC_CP(bbase + (size_t)(lr + 64) * KDIM    + lc, &ldsB[buf][(lr + 64) * LDP + lc]);
  };

  issueTile(0, 0);
  for (int kt = 0; kt < NKT; ++kt) {
    __builtin_amdgcn_s_wait_asynccnt(0);   // this wave's tile-kt copies landed in LDS
    __syncthreads();                       // everyone's copies landed; compute(kt-1) done
    if (kt + 1 < NKT) issueTile(kt + 1, (kt + 1) & 1);
    computeTile(kt & 1);
  }
#else
  // ---- fallback: register-prefetch through VGPRs into single LDS buffer ----
  auto fetchA = [&](int kt, v8bf& r0, v8bf& r1) {
    const __bf16* base; size_t stride;
    srcA(kt, base, stride);
    r0 = *(const v8bf*)(base + (size_t)lr        * stride + lc);
    r1 = *(const v8bf*)(base + (size_t)(lr + 64) * stride + lc);
  };
  auto fetchB = [&](int kt, v8bf& r0, v8bf& r1) {
    const __bf16* base = WT + (size_t)nBase * KDIM + kt * BK;
    r0 = *(const v8bf*)(base + (size_t)lr        * KDIM + lc);
    r1 = *(const v8bf*)(base + (size_t)(lr + 64) * KDIM + lc);
  };

  v8bf pa0, pa1, pb0, pb1;
  fetchA(0, pa0, pa1);
  fetchB(0, pb0, pb1);
  for (int kt = 0; kt < NKT; ++kt) {
    __syncthreads();
    *(v8bf*)(&ldsA[0][lr        * LDP + lc]) = pa0;
    *(v8bf*)(&ldsA[0][(lr + 64) * LDP + lc]) = pa1;
    *(v8bf*)(&ldsB[0][lr        * LDP + lc]) = pb0;
    *(v8bf*)(&ldsB[0][(lr + 64) * LDP + lc]) = pb1;
    __syncthreads();
    if (kt + 1 < NKT) { fetchA(kt + 1, pa0, pa1); fetchB(kt + 1, pb0, pb1); }
    computeTile(0);
  }
#endif

  // epilogue — C layout: VGPR i, lanes0-15 -> (M=i, N=lane); lanes16-31 -> (M=8+i, N=lane-16)
  const int gm0 = mBase + wr * 64 + (lane >> 4) * 8;
  const int gn0 = nBase + wc * 32 + l16;
#pragma unroll
  for (int mt = 0; mt < 4; ++mt)
#pragma unroll
    for (int nt = 0; nt < 2; ++nt)
#pragma unroll
      for (int i = 0; i < 8; ++i) {
        int gm = gm0 + mt * 16 + i;
        int gn = gn0 + nt * 16;
        size_t off = (size_t)gm * NDIM + gn;
        float v = acc[mt][nt][i];
        if (GATE == 0) {
          if (blockIdx.z == 0) zout[off] = sigmoidf_(v);                       // z gate
          else RH_out[off] = (__bf16)(sigmoidf_(v) * h_prev[off]);             // r*h
        } else {
          float z  = zout[off];                                                // z written in phase 1
          float hh = tanhf(v);
          zout[off] = z * h_prev[off] + (1.0f - z) * hh;                       // final h
        }
      }
}

extern "C" void kernel_launch(void* const* d_in, const int* in_sizes, int n_in,
                              void* d_out, int out_size, void* d_ws, size_t ws_size,
                              hipStream_t stream) {
  (void)in_sizes; (void)n_in; (void)out_size; (void)ws_size;
  const float* x  = (const float*)d_in[0];
  const float* h  = (const float*)d_in[1];
  const float* Wz = (const float*)d_in[2];
  const float* Wr = (const float*)d_in[3];
  const float* Wh = (const float*)d_in[4];
  float* out = (float*)d_out;

  char* ws = (char*)d_ws;
  const size_t szA = (size_t)BDIM * KDIM * 2;        // 32 MB bf16 x_h
  const size_t szW = (size_t)NDIM * KDIM * 2;        // 1 MB per transposed weight
  __bf16* Abuf = (__bf16*)(ws);
  __bf16* WzT  = (__bf16*)(ws + szA);
  __bf16* WrT  = (__bf16*)(ws + szA + szW);
  __bf16* WhT  = (__bf16*)(ws + szA + 2 * szW);
  __bf16* RH   = (__bf16*)(ws + szA + 3 * szW);      // 16 MB bf16 r*h

  pack_xh<<<dim3((BDIM * KDIM) / (8 * 256)), dim3(256), 0, stream>>>(x, h, Abuf);
  conv_wt<<<dim3((KDIM * NDIM) / 256, 3), dim3(256), 0, stream>>>(Wz, Wr, Wh, WzT, WrT, WhT);

  // phase 1: z and r gates (z -> d_out as scratch, r*h -> RH)
  gru_gemm<0><<<dim3(NDIM / 128, BDIM / 128, 2), dim3(256), 0, stream>>>(
      Abuf, nullptr, WzT, WrT, h, out, RH);
  // phase 2: h_hat + final blend into d_out
  gru_gemm<2><<<dim3(NDIM / 128, BDIM / 128, 1), dim3(256), 0, stream>>>(
      Abuf, RH, WhT, nullptr, h, out, nullptr);
}
